// Decoder_57458072486168
// MI455X (gfx1250) — matrix operand order repaired and checked by reading, not compiled, not run on hardware
//
#include <hip/hip_runtime.h>
#include <hip/hip_bf16.h>
#include <math.h>

// ---------------- problem constants (match reference setup_inputs) ----------
#define B_    32
#define H_    512
#define G_    1536          // 3*H
#define V_    32000
#define NT_V  (V_/16)       // 2000 column tiles in output projection
#define MAXN_ 64
#define SOS_  1

typedef __attribute__((ext_vector_type(16))) __bf16 v16bf;
typedef __attribute__((ext_vector_type(8)))  float  v8f;

// ---------------- tile load helpers (wave32 WMMA VGPR layouts) --------------
// B tile (32x16 bf16, K-major source = weight row W[n][k] contiguous in k):
//   lane l: n = l&15, holds K = 16*(l>>4) + e, e=0..15  -> one contiguous 32B run
__device__ __forceinline__ v16bf ld_b16_contig(const __bf16* p) {
    union { uint4 q[2]; v16bf v; } u;
    const uint4* s = (const uint4*)p;
    u.q[0] = s[0]; u.q[1] = s[1];
    return u.v;
}
// A tile (16x32 bf16, row m contiguous in k):
//   lane l: m = l&15, runs k0 = 8*(l>>4)+0..7 and 16+8*(l>>4)+0..7
__device__ __forceinline__ v16bf ld_b16_split(const __bf16* p0, const __bf16* p1) {
    union { uint4 q[2]; v16bf v; } u;
    u.q[0] = *(const uint4*)p0;
    u.q[1] = *(const uint4*)p1;
    return u.v;
}
// same A-tile layout but converting fp32 source (embedding gather) on the fly
__device__ __forceinline__ v16bf cvt_f32_split(const float* p0, const float* p1) {
    const float4* a = (const float4*)p0;
    const float4* b = (const float4*)p1;
    float4 a0 = a[0], a1 = a[1], b0 = b[0], b1 = b[1];
    v16bf r;
    r[0]=(__bf16)a0.x; r[1]=(__bf16)a0.y; r[2]=(__bf16)a0.z; r[3]=(__bf16)a0.w;
    r[4]=(__bf16)a1.x; r[5]=(__bf16)a1.y; r[6]=(__bf16)a1.z; r[7]=(__bf16)a1.w;
    r[8]=(__bf16)b0.x; r[9]=(__bf16)b0.y; r[10]=(__bf16)b0.z; r[11]=(__bf16)b0.w;
    r[12]=(__bf16)b1.x; r[13]=(__bf16)b1.y; r[14]=(__bf16)b1.z; r[15]=(__bf16)b1.w;
    return r;
}

__device__ __forceinline__ v8f wmma_bf16(v16bf a, v16bf b, v8f c) {
    return __builtin_amdgcn_wmma_f32_16x16x32_bf16(
        /*neg_a=*/false, a, /*neg_b=*/false, b,
        /*c_mod=*/(short)0, c, /*reuse_a=*/false, /*reuse_b=*/false);
}

// ---------------- prologue kernels ------------------------------------------
__global__ __launch_bounds__(256) void f32_to_bf16_k(const float* __restrict__ src,
                                                     __bf16* __restrict__ dst, int n) {
    int i = blockIdx.x * 256 + threadIdx.x;
    if (i < n) dst[i] = (__bf16)src[i];
}

__global__ __launch_bounds__(256) void init_state_k(const float* __restrict__ h_in,
                                                    float* h0, float* h1,
                                                    __bf16* h0b, __bf16* h1b,
                                                    int* tok) {
    int i = blockIdx.x * 256 + threadIdx.x;          // 0 .. B*H-1
    if (blockIdx.x == 0 && threadIdx.x < B_) tok[threadIdx.x] = SOS_;
    if (i < B_ * H_) {
        float a = h_in[i];                           // layer 0 state
        float c = h_in[B_ * H_ + i];                 // layer 1 state
        h0[i] = a; h0b[i] = (__bf16)a;
        h1[i] = c; h1b[i] = (__bf16)c;
    }
}

// ---------------- GRU gate GEMM:  out[z] = A[z] @ W[z]^T + bias[z] ----------
// grid = (G_/16, 2); z=0 -> gi (A = emb[tok] gathered, or x bf16),
//                    z=1 -> gh (A = h bf16).  One wave per 16-col tile, both M halves.
__global__ __launch_bounds__(32) void gemm_gates_k(
    const __bf16* __restrict__ A0, const __bf16* __restrict__ A1,
    const float*  __restrict__ embW, const int* __restrict__ tok,
    const __bf16* __restrict__ B0, const __bf16* __restrict__ B1,
    const float*  __restrict__ bias0, const float* __restrict__ bias1,
    float* __restrict__ out0, float* __restrict__ out1)
{
    const int nt   = blockIdx.x;          // 0..95
    const int z    = blockIdx.y;          // 0: gi, 1: gh
    const int lane = threadIdx.x;
    const int m    = lane & 15;
    const int half = lane >> 4;

    const __bf16* Bw   = z ? B1 : B0;
    const __bf16* Abf  = z ? A1 : A0;
    const float*  bias = z ? bias1 : bias0;
    float*        out  = z ? out1 : out0;
    const bool gather  = (Abf == nullptr);   // embedding-gather + convert path

    const float*  ar0f = nullptr; const float*  ar1f = nullptr;
    const __bf16* ar0b = nullptr; const __bf16* ar1b = nullptr;
    if (gather) {
        ar0f = embW + (size_t)tok[m]      * H_;
        ar1f = embW + (size_t)tok[m + 16] * H_;
    } else {
        ar0b = Abf + (size_t)m        * H_;
        ar1b = Abf + (size_t)(m + 16) * H_;
    }
    const __bf16* brow = Bw + (size_t)(nt * 16 + m) * H_;

    v8f acc0 = {}; v8f acc1 = {};
    #pragma unroll 4
    for (int kk = 0; kk < H_ / 32; ++kk) {
        const int kb = kk * 32;
        v16bf bt = ld_b16_split(brow + kb + 8 * half, brow + kb + 8 * half + 16);
        v16bf a0, a1;
        const int ka = kb + 8 * half;
        if (gather) {
            a0 = cvt_f32_split(ar0f + ka, ar0f + ka + 16);
            a1 = cvt_f32_split(ar1f + ka, ar1f + ka + 16);
        } else {
            a0 = ld_b16_split(ar0b + ka, ar0b + ka + 16);
            a1 = ld_b16_split(ar1b + ka, ar1b + ka + 16);
        }
        acc0 = wmma_bf16(a0, bt, acc0);
        acc1 = wmma_bf16(a1, bt, acc1);
    }

    const int   col = nt * 16 + m;        // D tile: lane col = lane&15
    const float bv  = bias[col];
    #pragma unroll
    for (int j = 0; j < 8; ++j) {         // D tile: lane row = j + 8*half
        int r = j + 8 * half;
        out[(size_t)r        * G_ + col] = acc0[j] + bv;
        out[(size_t)(r + 16) * G_ + col] = acc1[j] + bv;
    }
}

// ---------------- GRU pointwise gates ---------------------------------------
__global__ __launch_bounds__(256) void gru_gates_k(
    const float* __restrict__ gi, const float* __restrict__ gh,
    float* __restrict__ h, __bf16* __restrict__ hb)
{
    int i = blockIdx.x * 256 + threadIdx.x;          // 0..B*H-1
    if (i >= B_ * H_) return;
    int b = i >> 9, j = i & (H_ - 1);
    const float* gib = gi + (size_t)b * G_;
    const float* ghb = gh + (size_t)b * G_;
    float r = 1.f / (1.f + expf(-(gib[j]         + ghb[j])));
    float z = 1.f / (1.f + expf(-(gib[j + H_]    + ghb[j + H_])));
    float n = tanhf(gib[j + 2 * H_] + r * ghb[j + 2 * H_]);
    float hn = (1.f - z) * n + z * h[i];
    h[i]  = hn;
    hb[i] = (__bf16)hn;
}

// ---------------- logits GEMM + per-tile argmax -----------------------------
// grid = NT_V (=2000) waves; each wave: 16 output cols x full M=32, K=512,
// then cross-lane argmax (width-16, matching D-tile row placement).
__global__ __launch_bounds__(32) void logits_argmax_k(
    const __bf16* __restrict__ x, const __bf16* __restrict__ Wo,
    const float* __restrict__ bout,
    float* __restrict__ pmax, int* __restrict__ pidx)
{
    const int nt   = blockIdx.x;
    const int lane = threadIdx.x;
    const int m    = lane & 15;
    const int half = lane >> 4;

    const __bf16* ar0  = x + (size_t)m        * H_;
    const __bf16* ar1  = x + (size_t)(m + 16) * H_;
    const __bf16* brow = Wo + (size_t)(nt * 16 + m) * H_;

    v8f acc[2] = {{}, {}};
    #pragma unroll 4
    for (int kk = 0; kk < H_ / 32; ++kk) {
        if (kk + 2 < H_ / 32)
            __builtin_prefetch(brow + (kk + 2) * 32 + 16 * half, 0, 1);
        const int kb = kk * 32;
        v16bf bt = ld_b16_split(brow + kb + 8 * half, brow + kb + 8 * half + 16);
        const int ka = kb + 8 * half;
        v16bf a0 = ld_b16_split(ar0 + ka, ar0 + ka + 16);
        v16bf a1 = ld_b16_split(ar1 + ka, ar1 + ka + 16);
        acc[0] = wmma_bf16(a0, bt, acc[0]);
        acc[1] = wmma_bf16(a1, bt, acc[1]);
    }

    const float bv = bout[nt * 16 + m];
    #pragma unroll
    for (int t = 0; t < 2; ++t) {
        #pragma unroll
        for (int j = 0; j < 8; ++j) {
            float v  = acc[t][j] + bv;
            int   id = nt * 16 + m;
            // argmax across the 16 lanes that hold this row (first-max tie rule)
            #pragma unroll
            for (int off = 8; off; off >>= 1) {
                float ov = __shfl_xor(v,  off, 16);
                int   oi = __shfl_xor(id, off, 16);
                if (ov > v || (ov == v && oi < id)) { v = ov; id = oi; }
            }
            if (m == 0) {
                int row = t * 16 + j + 8 * half;
                pmax[(size_t)row * NT_V + nt] = v;
                pidx[(size_t)row * NT_V + nt] = id;
            }
        }
    }
}

// ---------------- final argmax reduce + token emit --------------------------
__global__ __launch_bounds__(256) void argmax_reduce_k(
    const float* __restrict__ pmax, const int* __restrict__ pidx,
    int* __restrict__ tok, float* __restrict__ toks_out, int step)
{
    const int b = blockIdx.x, t = threadIdx.x;
    float v = -3.4e38f; int id = 0x7fffffff;
    for (int c = t; c < NT_V; c += 256) {
        float pv = pmax[(size_t)b * NT_V + c];
        int   pi = pidx[(size_t)b * NT_V + c];
        if (pv > v || (pv == v && pi < id)) { v = pv; id = pi; }
    }
    __shared__ float sv[256];
    __shared__ int   si[256];
    sv[t] = v; si[t] = id;
    __syncthreads();
    for (int s = 128; s > 0; s >>= 1) {
        if (t < s) {
            if (sv[t + s] > sv[t] || (sv[t + s] == sv[t] && si[t + s] < si[t])) {
                sv[t] = sv[t + s]; si[t] = si[t + s];
            }
        }
        __syncthreads();
    }
    if (t == 0) {
        tok[b] = si[0];
        toks_out[(size_t)b * MAXN_ + step] = (float)si[0];
    }
}

// ---------------- epilogue: h_t ---------------------------------------------
__global__ __launch_bounds__(256) void write_hidden_k(const float* __restrict__ h0,
                                                      const float* __restrict__ h1,
                                                      float* __restrict__ out) {
    int i = blockIdx.x * 256 + threadIdx.x;          // 0 .. 2*B*H-1
    if (i < 2 * B_ * H_)
        out[i] = (i < B_ * H_) ? h0[i] : h1[i - B_ * H_];
}

// ---------------- host entry ------------------------------------------------
extern "C" void kernel_launch(void* const* d_in, const int* in_sizes, int n_in,
                              void* d_out, int out_size, void* d_ws, size_t ws_size,
                              hipStream_t stream)
{
    (void)in_sizes; (void)n_in; (void)out_size; (void)ws_size;
    const float* h_in  = (const float*)d_in[0];
    const float* emb   = (const float*)d_in[1];
    const float* Wih0  = (const float*)d_in[2];
    const float* Whh0  = (const float*)d_in[3];
    const float* bih0  = (const float*)d_in[4];
    const float* bhh0  = (const float*)d_in[5];
    const float* Wih1  = (const float*)d_in[6];
    const float* Whh1  = (const float*)d_in[7];
    const float* bih1  = (const float*)d_in[8];
    const float* bhh1  = (const float*)d_in[9];
    const float* Wout  = (const float*)d_in[10];
    const float* bout  = (const float*)d_in[11];
    float* out = (float*)d_out;

    // --- workspace carve-out (256B aligned), ~40 MB total ---
    char* w = (char*)d_ws;
    size_t off = 0;
    auto take = [&](size_t bytes) { size_t o = off; off += (bytes + 255) & ~(size_t)255; return o; };
    __bf16* Wih0b = (__bf16*)(w + take((size_t)G_ * H_ * 2));
    __bf16* Whh0b = (__bf16*)(w + take((size_t)G_ * H_ * 2));
    __bf16* Wih1b = (__bf16*)(w + take((size_t)G_ * H_ * 2));
    __bf16* Whh1b = (__bf16*)(w + take((size_t)G_ * H_ * 2));
    __bf16* Woutb = (__bf16*)(w + take((size_t)V_ * H_ * 2));
    float*  gi    = (float*)(w + take((size_t)B_ * G_ * 4));
    float*  gh    = (float*)(w + take((size_t)B_ * G_ * 4));
    float*  h0    = (float*)(w + take((size_t)B_ * H_ * 4));
    float*  h1    = (float*)(w + take((size_t)B_ * H_ * 4));
    __bf16* h0b   = (__bf16*)(w + take((size_t)B_ * H_ * 2));
    __bf16* h1b   = (__bf16*)(w + take((size_t)B_ * H_ * 2));
    float*  pmax  = (float*)(w + take((size_t)B_ * NT_V * 4));
    int*    pidx  = (int*)  (w + take((size_t)B_ * NT_V * 4));
    int*    tok   = (int*)  (w + take(256));

    // --- one-time: fp32 -> bf16 weight conversion (lives in L2 thereafter) ---
    const int nW = G_ * H_;
    f32_to_bf16_k<<<(nW + 255) / 256, 256, 0, stream>>>(Wih0, Wih0b, nW);
    f32_to_bf16_k<<<(nW + 255) / 256, 256, 0, stream>>>(Whh0, Whh0b, nW);
    f32_to_bf16_k<<<(nW + 255) / 256, 256, 0, stream>>>(Wih1, Wih1b, nW);
    f32_to_bf16_k<<<(nW + 255) / 256, 256, 0, stream>>>(Whh1, Whh1b, nW);
    const int nWo = V_ * H_;
    f32_to_bf16_k<<<(nWo + 255) / 256, 256, 0, stream>>>(Wout, Woutb, nWo);
    init_state_k<<<(B_ * H_ + 255) / 256, 256, 0, stream>>>(h_in, h0, h1, h0b, h1b, tok);

    // --- autoregressive decode loop ---
    const dim3 gg(G_ / 16, 2);
    for (int step = 0; step < MAXN_; ++step) {
        // layer 0: gi = emb[tok] @ Wih0^T + bih0 ; gh = h0 @ Whh0^T + bhh0
        gemm_gates_k<<<gg, 32, 0, stream>>>(nullptr, h0b, emb, tok,
                                            Wih0b, Whh0b, bih0, bhh0, gi, gh);
        gru_gates_k<<<(B_ * H_ + 255) / 256, 256, 0, stream>>>(gi, gh, h0, h0b);
        // layer 1: gi = h0_new @ Wih1^T + bih1 ; gh = h1 @ Whh1^T + bhh1
        gemm_gates_k<<<gg, 32, 0, stream>>>(h0b, h1b, nullptr, nullptr,
                                            Wih1b, Whh1b, bih1, bhh1, gi, gh);
        gru_gates_k<<<(B_ * H_ + 255) / 256, 256, 0, stream>>>(gi, gh, h1, h1b);
        // output projection + argmax
        logits_argmax_k<<<NT_V, 32, 0, stream>>>(h1b, Woutb, bout, pmax, pidx);
        argmax_reduce_k<<<B_, 256, 0, stream>>>(pmax, pidx, tok, out, step);
    }
    // h_t = [2, B, H]
    write_hidden_k<<<(2 * B_ * H_ + 255) / 256, 256, 0, stream>>>(h0, h1, out + B_ * MAXN_);
}
